// MethodEncoder_90555090469433
// MI455X (gfx1250) — matrix operand order repaired and checked by reading, not compiled, not run on hardware
//
#include <hip/hip_runtime.h>
#include <hip/hip_bf16.h>

// ---------------------------------------------------------------------------
// GGNN encoder for MI455X (gfx1250, wave32, WMMA).
// GEMMs run on v_wmma_f32_16x16x32_bf16 (f32 accumulate); edge aggregation is
// f32 global atomics (L2-resident: h/m/agg ~30MB each << 192MB L2).
// f32->bf16 conversion goes through __builtin_convertvector so ISel can use
// the native packed bf16 convert when the target provides it.
// Workspace layout (floats):
//   [0,NH)      h0        NH = 30000*256
//   [NH,2NH)    h1        (double buffer for GRU output)
//   [2NH,3NH)   agg
//   [3NH,4NH)   m
//   [4NH,+16K)  pooled (64*256)
//   then        counts (64)
// Total ~123 MB.
// ---------------------------------------------------------------------------

#define NNODES 30000
#define NEDGES 480000
#define NGRAPHS 64
#define IN_DIM 384
#define HID 256
#define STEPS 5
#define LN_EPS 1e-5f

typedef __attribute__((ext_vector_type(16))) __bf16 v16bf;
typedef __attribute__((ext_vector_type(8)))  float  v8f;
typedef __attribute__((ext_vector_type(2)))  __bf16 v2bf;
typedef __attribute__((ext_vector_type(2)))  float  v2f;

union FragBF { v16bf v; uint4 q[2]; };

// store two bf16 (4 bytes); d must be 4B-aligned (all call sites are).
// __builtin_convertvector -> fptrunc f32->bf16; backend selects the native
// (packed) convert instruction when available.
__device__ __forceinline__ void st_bf2(unsigned short* d, float x, float y) {
    v2f s; s.x = x; s.y = y;
    *(v2bf*)d = __builtin_convertvector(s, v2bf);
}

__device__ __forceinline__ float sigmoidf_(float x) { return 1.f / (1.f + __expf(-x)); }

__device__ __forceinline__ v8f wmma_bf16(const FragBF& a, const FragBF& b, v8f c) {
    return __builtin_amdgcn_wmma_f32_16x16x32_bf16(false, a.v, false, b.v,
                                                   (short)0, c, false, false);
}

__device__ __forceinline__ v8f vzero8() {
    v8f z = {0.f, 0.f, 0.f, 0.f, 0.f, 0.f, 0.f, 0.f};
    return z;
}

// ---------------------------------------------------------------------------
// Generic GEMM: C[M,N] = A[M,K] * B (+bias) (+relu). bf16 WMMA, f32 accum.
// transB=1: B stored [N,K] row-major (access B[n*ldb+k]); else B is [K,N].
// zbuf (optional, same shape as C) is zeroed in the epilogue (used to reset
// `agg` just before the scatter without racing the previous GRU kernel).
// Block: 256 thr (8 waves). Tile 64x128, wave tile 16x64 (4 wmma accums).
// ---------------------------------------------------------------------------
__global__ __launch_bounds__(256)
void gemm_bf16_kernel(const float* __restrict__ A, const float* __restrict__ B,
                      const float* __restrict__ bias, float* __restrict__ C,
                      float* __restrict__ zbuf,
                      int M, int K, int N, int ldb, int transB, int relu)
{
    __shared__ unsigned short lds_a[64 * 40];    // [row][k], pad 40
    __shared__ unsigned short lds_bt[128 * 40];  // [n][k],  pad 40

    const int tid  = threadIdx.x;
    const int lane = tid & 31, wave = tid >> 5;
    const int wr = wave >> 1;            // 0..3 : 16-row group
    const int wc = wave & 1;             // 0..1 : 64-col group
    const int mlo = lane & 15, g = lane >> 4;
    const int row0 = blockIdx.y * 64;
    const int col0 = blockIdx.x * 128;

    v8f acc[4] = {vzero8(), vzero8(), vzero8(), vzero8()};

    for (int kk = 0; kk < K; kk += 32) {
        // ---- A tile 64x32 -> LDS (bf16) ----
        {
            int r  = tid >> 3;
            int c4 = (tid & 7) * 4;
#pragma unroll
            for (int p = 0; p < 2; ++p) {
                int rr  = r + p * 32;
                int row = row0 + rr; if (row > M - 1) row = M - 1;
                float4 va = *(const float4*)(A + (size_t)row * K + kk + c4);
                unsigned short* d = &lds_a[rr * 40 + c4];
                st_bf2(d,     va.x, va.y);
                st_bf2(d + 2, va.z, va.w);
            }
        }
        // ---- B tile -> LDS as BT[n][k] (bf16) ----
        if (transB) {
            int n  = tid >> 3;
            int c4 = (tid & 7) * 4;
#pragma unroll
            for (int p = 0; p < 4; ++p) {
                int nn = n + p * 32;
                float4 vb = *(const float4*)(B + (size_t)(col0 + nn) * ldb + kk + c4);
                unsigned short* d = &lds_bt[nn * 40 + c4];
                st_bf2(d,     vb.x, vb.y);
                st_bf2(d + 2, vb.z, vb.w);
            }
        } else {
            // read two adjacent K-rows so bf16 pairs pack along K
            int k2 = (tid >> 5) * 2;          // 0,2,...,14
            int n4 = (tid & 31) * 4;
#pragma unroll
            for (int p = 0; p < 2; ++p) {
                int kx = k2 + p * 16;
                float4 v0 = *(const float4*)(B + (size_t)(kk + kx)     * ldb + col0 + n4);
                float4 v1 = *(const float4*)(B + (size_t)(kk + kx + 1) * ldb + col0 + n4);
                st_bf2(&lds_bt[(n4 + 0) * 40 + kx], v0.x, v1.x);
                st_bf2(&lds_bt[(n4 + 1) * 40 + kx], v0.y, v1.y);
                st_bf2(&lds_bt[(n4 + 2) * 40 + kx], v0.z, v1.z);
                st_bf2(&lds_bt[(n4 + 3) * 40 + kx], v0.w, v1.w);
            }
        }
        __syncthreads();

        // ---- fragments + WMMA ----
        FragBF fa;
        const unsigned short* pa = &lds_a[(wr * 16 + mlo) * 40 + 8 * g];
        fa.q[0] = *(const uint4*)pa;          // K = 8g .. 8g+7
        fa.q[1] = *(const uint4*)(pa + 16);   // K = 16+8g .. 23+8g
#pragma unroll
        for (int nb = 0; nb < 4; ++nb) {
            FragBF fb;
            const unsigned short* pb = &lds_bt[(wc * 64 + nb * 16 + mlo) * 40 + 8 * g];
            fb.q[0] = *(const uint4*)pb;      // K = 8g .. 8g+7
            fb.q[1] = *(const uint4*)(pb + 16);// K = 16+8g .. 23+8g
            acc[nb] = wmma_bf16(fa, fb, acc[nb]);
        }
        __syncthreads();
    }

    // ---- epilogue: D[r+8g][n] lives in vgpr r of lane 16g+n ----
#pragma unroll
    for (int nb = 0; nb < 4; ++nb) {
        int n = col0 + wc * 64 + nb * 16 + mlo;
        float bv = bias ? bias[n] : 0.f;
#pragma unroll
        for (int r = 0; r < 8; ++r) {
            int row = row0 + wr * 16 + r + 8 * g;
            if (row < M) {
                float v = acc[nb][r] + bv;
                if (relu) v = fmaxf(v, 0.f);
                C[(size_t)row * N + n] = v;
                if (zbuf) zbuf[(size_t)row * N + n] = 0.f;
            }
        }
    }
}

// ---------------------------------------------------------------------------
// Fused GRU step: for each node row,
//   gi = agg @ W_ih^T + b_ih ; gh = h @ W_hh^T + b_hh  (3 gates each, K=256)
//   r = sig(ir+hr); z = sig(iz+hz); n = tanh(in + r*hn); h' = (1-z)n + z h
// Block tile: 64 rows x 64 hid. 8 waves: wr(0..1) x wc(0..3); each wave owns
// 32 rows x 16 hid and carries 12 wmma accumulators (2 rowhalves x 3 gates x
// {i,h}). Writes Hout (double buffer - no in-place race with readers of Hin).
// ---------------------------------------------------------------------------
__global__ __launch_bounds__(256)
void gru_fused_kernel(const float* __restrict__ Agg, const float* __restrict__ Hin,
                      const float* __restrict__ Wih, const float* __restrict__ Whh,
                      const float* __restrict__ bih, const float* __restrict__ bhh,
                      float* __restrict__ Hout, int M)
{
    __shared__ unsigned short lds_aa[64 * 40];       // agg rows
    __shared__ unsigned short lds_ah[64 * 40];       // h rows
    __shared__ unsigned short lds_bt[6 * 64 * 40];   // [mat*3+gate][n][k]

    const int tid  = threadIdx.x;
    const int lane = tid & 31, wave = tid >> 5;
    const int wr = wave >> 2;          // 0..1 : 32-row half
    const int wc = wave & 3;           // 0..3 : 16-hid group
    const int mlo = lane & 15, g = lane >> 4;
    const int row0 = blockIdx.y * 64;
    const int hid0 = blockIdx.x * 64;

    v8f acc_i[2][3], acc_h[2][3];
#pragma unroll
    for (int a = 0; a < 2; ++a)
#pragma unroll
        for (int b = 0; b < 3; ++b) { acc_i[a][b] = vzero8(); acc_h[a][b] = vzero8(); }

    for (int kk = 0; kk < HID; kk += 32) {
        // ---- A tiles (agg + h), 64x32 each ----
        {
            int r  = tid >> 3;
            int c4 = (tid & 7) * 4;
#pragma unroll
            for (int p = 0; p < 2; ++p) {
                int rr  = r + p * 32;
                int row = row0 + rr; if (row > M - 1) row = M - 1;
                float4 va = *(const float4*)(Agg + (size_t)row * HID + kk + c4);
                float4 vh = *(const float4*)(Hin + (size_t)row * HID + kk + c4);
                unsigned short* da = &lds_aa[rr * 40 + c4];
                st_bf2(da,     va.x, va.y);
                st_bf2(da + 2, va.z, va.w);
                unsigned short* dh = &lds_ah[rr * 40 + c4];
                st_bf2(dh,     vh.x, vh.y);
                st_bf2(dh + 2, vh.z, vh.w);
            }
        }
        // ---- B tiles: W_ih / W_hh rows (gate*256 + hid0 + n), K slice ----
        {
            int n  = tid >> 3;
            int c4 = (tid & 7) * 4;
#pragma unroll
            for (int mat = 0; mat < 2; ++mat) {
                const float* W = mat ? Whh : Wih;
#pragma unroll
                for (int gt = 0; gt < 3; ++gt) {
#pragma unroll
                    for (int p = 0; p < 2; ++p) {
                        int nn = n + p * 32;
                        int wrow = gt * HID + hid0 + nn;
                        float4 vb = *(const float4*)(W + (size_t)wrow * HID + kk + c4);
                        unsigned short* d = &lds_bt[((mat * 3 + gt) * 64 + nn) * 40 + c4];
                        st_bf2(d,     vb.x, vb.y);
                        st_bf2(d + 2, vb.z, vb.w);
                    }
                }
            }
        }
        __syncthreads();

        FragBF faa[2], fah[2], fb[6];
#pragma unroll
        for (int hh = 0; hh < 2; ++hh) {
            const unsigned short* pa = &lds_aa[(wr * 32 + hh * 16 + mlo) * 40 + 8 * g];
            faa[hh].q[0] = *(const uint4*)pa;  faa[hh].q[1] = *(const uint4*)(pa + 16);
            const unsigned short* ph = &lds_ah[(wr * 32 + hh * 16 + mlo) * 40 + 8 * g];
            fah[hh].q[0] = *(const uint4*)ph;  fah[hh].q[1] = *(const uint4*)(ph + 16);
        }
#pragma unroll
        for (int f = 0; f < 6; ++f) {
            const unsigned short* pb = &lds_bt[(f * 64 + wc * 16 + mlo) * 40 + 8 * g];
            fb[f].q[0] = *(const uint4*)pb;    fb[f].q[1] = *(const uint4*)(pb + 16);
        }
#pragma unroll
        for (int hh = 0; hh < 2; ++hh)
#pragma unroll
            for (int gt = 0; gt < 3; ++gt) {
                acc_i[hh][gt] = wmma_bf16(faa[hh], fb[gt],     acc_i[hh][gt]);
                acc_h[hh][gt] = wmma_bf16(fah[hh], fb[3 + gt], acc_h[hh][gt]);
            }
        __syncthreads();
    }

    // ---- gate epilogue ----
    const int hc = hid0 + wc * 16 + mlo;
    const float bir = bih[hc], biz = bih[hc + HID], bin_ = bih[hc + 2 * HID];
    const float bhr = bhh[hc], bhz = bhh[hc + HID], bhn  = bhh[hc + 2 * HID];
#pragma unroll
    for (int hh = 0; hh < 2; ++hh) {
#pragma unroll
        for (int r = 0; r < 8; ++r) {
            int row = row0 + wr * 32 + hh * 16 + r + 8 * g;
            if (row < M) {
                float ir  = acc_i[hh][0][r] + bir;
                float iz  = acc_i[hh][1][r] + biz;
                float inn = acc_i[hh][2][r] + bin_;
                float hr  = acc_h[hh][0][r] + bhr;
                float hz  = acc_h[hh][1][r] + bhz;
                float hn  = acc_h[hh][2][r] + bhn;
                float rg = sigmoidf_(ir + hr);
                float zg = sigmoidf_(iz + hz);
                float ng = tanhf(inn + rg * hn);
                float hv = Hin[(size_t)row * HID + hc];
                Hout[(size_t)row * HID + hc] = (1.f - zg) * ng + zg * hv;
            }
        }
    }
}

// ---------------------------------------------------------------------------
// LayerNorm over HID=256 per row, one wave per row (8 floats/lane), in place.
// ---------------------------------------------------------------------------
__global__ __launch_bounds__(256)
void ln_kernel(float* __restrict__ H, const float* __restrict__ gamma,
               const float* __restrict__ beta, int M)
{
    int wave = threadIdx.x >> 5, lane = threadIdx.x & 31;
    int row = blockIdx.x * 8 + wave;
    if (row >= M) return;
    float* p = H + (size_t)row * HID + lane * 8;
    float4 a = *(const float4*)p;
    float4 b = *(const float4*)(p + 4);
    float s = a.x + a.y + a.z + a.w + b.x + b.y + b.z + b.w;
    float q = a.x * a.x + a.y * a.y + a.z * a.z + a.w * a.w +
              b.x * b.x + b.y * b.y + b.z * b.z + b.w * b.w;
#pragma unroll
    for (int off = 16; off > 0; off >>= 1) {
        s += __shfl_xor(s, off, 32);
        q += __shfl_xor(q, off, 32);
    }
    float mu = s * (1.f / HID);
    float var = q * (1.f / HID) - mu * mu;
    float rs = rsqrtf(var + LN_EPS);
    float4 g0 = *(const float4*)(gamma + lane * 8);
    float4 g1 = *(const float4*)(gamma + lane * 8 + 4);
    float4 b0 = *(const float4*)(beta + lane * 8);
    float4 b1 = *(const float4*)(beta + lane * 8 + 4);
    a.x = (a.x - mu) * rs * g0.x + b0.x;  a.y = (a.y - mu) * rs * g0.y + b0.y;
    a.z = (a.z - mu) * rs * g0.z + b0.z;  a.w = (a.w - mu) * rs * g0.w + b0.w;
    b.x = (b.x - mu) * rs * g1.x + b1.x;  b.y = (b.y - mu) * rs * g1.y + b1.y;
    b.z = (b.z - mu) * rs * g1.z + b1.z;  b.w = (b.w - mu) * rs * g1.w + b1.w;
    *(float4*)p = a;
    *(float4*)(p + 4) = b;
}

// ---------------------------------------------------------------------------
// Edge scatter-add: agg[dst] += m[src]. 4 edges/block, 4 channels/thread.
// ---------------------------------------------------------------------------
__global__ __launch_bounds__(256)
void scatter_kernel(const float* __restrict__ m, const int* __restrict__ src,
                    const int* __restrict__ dst, float* __restrict__ agg, int nE)
{
    int e = blockIdx.x * 4 + (threadIdx.x >> 6);
    if (e >= nE) return;
    int c = (threadIdx.x & 63) * 4;
    int s = src[e], d = dst[e];
    float4 v = *(const float4*)(m + (size_t)s * HID + c);
    float* a = agg + (size_t)d * HID + c;
    atomicAdd(a + 0, v.x); atomicAdd(a + 1, v.y);
    atomicAdd(a + 2, v.z); atomicAdd(a + 3, v.w);
}

// ---------------------------------------------------------------------------
// Pooling
// ---------------------------------------------------------------------------
__global__ __launch_bounds__(256)
void zero_small_kernel(float* __restrict__ pooled, float* __restrict__ counts)
{
    int i = blockIdx.x * 256 + threadIdx.x;
    if (i < NGRAPHS * HID) pooled[i] = 0.f;
    if (i < NGRAPHS) counts[i] = 0.f;
}

__global__ __launch_bounds__(256)
void pool_accum_kernel(const float* __restrict__ h, const int* __restrict__ batch,
                       float* __restrict__ pooled, float* __restrict__ counts, int M)
{
    int idx = blockIdx.x * 256 + threadIdx.x;
    int node = idx >> 6;
    if (node >= M) return;
    int c = (idx & 63) * 4;
    int b = batch[node];
    float4 v = *(const float4*)(h + (size_t)node * HID + c);
    float* p = pooled + (size_t)b * HID + c;
    atomicAdd(p + 0, v.x); atomicAdd(p + 1, v.y);
    atomicAdd(p + 2, v.z); atomicAdd(p + 3, v.w);
    if (c == 0) atomicAdd(&counts[b], 1.f);
}

__global__ __launch_bounds__(256)
void pool_final_kernel(const float* __restrict__ pooled, const float* __restrict__ counts,
                       float* __restrict__ out)
{
    int i = blockIdx.x * 256 + threadIdx.x;
    int g = i >> 8;
    out[i] = pooled[i] / fmaxf(counts[g], 1.f);
}

// ---------------------------------------------------------------------------
extern "C" void kernel_launch(void* const* d_in, const int* in_sizes, int n_in,
                              void* d_out, int out_size, void* d_ws, size_t ws_size,
                              hipStream_t stream)
{
    const float* x      = (const float*)d_in[0];
    const int*   ei     = (const int*)  d_in[1];   // [2, NEDGES]
    const int*   batch  = (const int*)  d_in[2];
    const float* lin_w  = (const float*)d_in[3];   // [HID, IN_DIM]
    const float* lin_b  = (const float*)d_in[4];
    const float* gamma  = (const float*)d_in[5];
    const float* beta   = (const float*)d_in[6];
    const float* ggnn_w = (const float*)d_in[7];   // [STEPS, HID, HID]
    const float* w_ih   = (const float*)d_in[8];   // [3*HID, HID]
    const float* w_hh   = (const float*)d_in[9];
    const float* b_ih   = (const float*)d_in[10];
    const float* b_hh   = (const float*)d_in[11];
    float* out = (float*)d_out;

    const size_t NH = (size_t)NNODES * HID;
    float* ws     = (float*)d_ws;
    float* h0     = ws;
    float* h1     = ws + NH;
    float* agg    = ws + 2 * NH;
    float* mbuf   = ws + 3 * NH;
    float* pooled = ws + 4 * NH;
    float* counts = pooled + NGRAPHS * HID;

    dim3 blk(256);
    dim3 gemm_grid(HID / 128, (NNODES + 63) / 64);

    zero_small_kernel<<<dim3(65), blk, 0, stream>>>(pooled, counts);

    // h = relu(x @ lin_w^T + lin_b)  [transB: lin_w is [HID, IN_DIM]]
    gemm_bf16_kernel<<<gemm_grid, blk, 0, stream>>>(
        x, lin_w, lin_b, h0, nullptr, NNODES, IN_DIM, HID, IN_DIM, 1, 1);
    ln_kernel<<<dim3(NNODES / 8), blk, 0, stream>>>(h0, gamma, beta, NNODES);

    float* hc = h0;
    float* hn = h1;
    for (int i = 0; i < STEPS; ++i) {
        // m = h @ ggnn_w[i]  (B is [K,N]); epilogue zeroes agg for the scatter
        gemm_bf16_kernel<<<gemm_grid, blk, 0, stream>>>(
            hc, ggnn_w + (size_t)i * HID * HID, nullptr, mbuf, agg,
            NNODES, HID, HID, HID, 0, 0);
        // agg[dst] += m[src]
        scatter_kernel<<<dim3(NEDGES / 4), blk, 0, stream>>>(
            mbuf, ei, ei + NEDGES, agg, NEDGES);
        // GRU cell -> hn
        gru_fused_kernel<<<dim3(HID / 64, (NNODES + 63) / 64), blk, 0, stream>>>(
            agg, hc, w_ih, w_hh, b_ih, b_hh, hn, NNODES);
        float* t = hc; hc = hn; hn = t;
    }

    ln_kernel<<<dim3(NNODES / 8), blk, 0, stream>>>(hc, gamma, beta, NNODES);
    pool_accum_kernel<<<dim3((NNODES * 64) / 256), blk, 0, stream>>>(
        hc, batch, pooled, counts, NNODES);
    pool_final_kernel<<<dim3((NGRAPHS * HID) / 256), blk, 0, stream>>>(
        pooled, counts, out);
}
